// GatedMemoryModule_77644418777919
// MI455X (gfx1250) — matrix operand order repaired and checked by reading, not compiled, not run on hardware
//
#include <hip/hip_runtime.h>

// ---------------------------------------------------------------------------
// GatedMemoryModule on MI455X (gfx1250, wave32, WMMA 16x16x32 f16 -> f32)
// SEQ=100, BATCH=1024, DIM=512, N_MEM=32, TEMP=0.1
// All four GEMMs (scores, wq, gate, retrieved) now run on v_wmma.
// ---------------------------------------------------------------------------

#define S_   100
#define B_   1024
#define D_   512
#define M_   32
#define ROWS (S_*B_)            // 102400 flattened (t,b) rows
#define INV_TEMP 10.0f
#define NPART 100               // split-K partials for wq

typedef __attribute__((ext_vector_type(16))) _Float16 v16h;
typedef __attribute__((ext_vector_type(8)))  float    v8f;

union HFrag {
    v16h v;
    unsigned int u[8];
};

__device__ __forceinline__ unsigned int pkh(float a, float b) {
    union { _Float16 h[2]; unsigned int u; } z;
    z.h[0] = (_Float16)a;
    z.h[1] = (_Float16)b;
    return z.u;
}

__device__ __forceinline__ unsigned short h16(float x) {
    union { _Float16 h; unsigned short s; } z;
    z.h = (_Float16)x;
    return z.s;
}

// 16-bit A-matrix 16x32 layout: lane row = lane&15,
//   lanes 0-15 : VGPR j -> K = 2j (j<4) / 16+2(j-4) (j>=4) ; lanes16-31 : +8
__device__ __forceinline__ int a_k(int j, int lane) {
    return 2*j + ((j >= 4) ? 8 : 0) + ((lane & 16) ? 8 : 0);
}
// 16-bit B-matrix 32x16 layout: lane col = lane&15,
//   lanes 0-15 : VGPR j -> K = 2j (j<4) / 16+2(j-4) (j>=4) ; lanes16-31 : +8
__device__ __forceinline__ int b_k(int j, int lane) {
    return 2*j + ((j >= 4) ? 8 : 0) + ((lane & 16) ? 8 : 0);
}

// ---------------------------------------------------------------------------
// Kernel 1a: scores[r][m] = q[r,:] . mem[m,:]   (raw, temperature applied later)
// grid 1600 x 128 threads (4 waves, 16 rows per wave)
// ---------------------------------------------------------------------------
__global__ __launch_bounds__(128)
void k_scores(const float* __restrict__ q, const float* __restrict__ mem,
              float* __restrict__ scores)
{
    __shared__ unsigned short Mb[M_*520];          // 32 rows x 512 f16, stride 520
    for (int idx = threadIdx.x; idx < M_*D_; idx += 128) {
        int m = idx >> 9, d = idx & 511;
        Mb[m*520 + d] = h16(mem[m*D_ + d]);
    }
    __syncthreads();
    const unsigned int* Mb32 = (const unsigned int*)Mb;   // stride 260 dwords

    const int wid  = threadIdx.x >> 5, lane = threadIdx.x & 31;
    const int rbase = (blockIdx.x*4 + wid)*16;
    const int n16  = lane & 15;
    const int hi8  = (lane & 16) ? 8 : 0;
    const int arow = rbase + n16;

    v8f acc0 = {}, acc1 = {};
    for (int kc = 0; kc < 16; ++kc) {
        const int kb = kc*32;
        HFrag a, b0, b1;
#pragma unroll
        for (int j = 0; j < 8; ++j) {
            int ka = kb + a_k(j, lane);
            float2 p = *(const float2*)(q + (size_t)arow*D_ + ka);
            a.u[j] = pkh(p.x, p.y);
            int kbb = kb + b_k(j, lane);
            b0.u[j] = Mb32[n16*260        + (kbb >> 1)];
            b1.u[j] = Mb32[(16 + n16)*260 + (kbb >> 1)];
        }
        acc0 = __builtin_amdgcn_wmma_f32_16x16x32_f16(false, a.v, false, b0.v, (short)0, acc0, false, false);
        acc1 = __builtin_amdgcn_wmma_f32_16x16x32_f16(false, a.v, false, b1.v, (short)0, acc1, false, false);
    }
#pragma unroll
    for (int i = 0; i < 8; ++i) {
        int r = rbase + i + hi8;
        scores[(size_t)r*M_ + n16]      = acc0[i];
        scores[(size_t)r*M_ + 16 + n16] = acc1[i];
    }
}

// ---------------------------------------------------------------------------
// Kernel 1b: softmax over the t axis per (b,m), in place. grid 128 x 256.
// addr = t*32768 + tid  -> perfectly coalesced.
// ---------------------------------------------------------------------------
__global__ __launch_bounds__(256)
void k_softmax_t(float* __restrict__ s)
{
    const int tid = blockIdx.x*256 + threadIdx.x;       // 0..32767 = b*32+m
    float mx = -3.0e38f;
    for (int t = 0; t < S_; ++t)
        mx = fmaxf(mx, s[(size_t)t*32768 + tid]);
    float sum = 0.f;
    for (int t = 0; t < S_; ++t) {
        float e = __expf((s[(size_t)t*32768 + tid] - mx) * INV_TEMP);
        s[(size_t)t*32768 + tid] = e;
        sum += e;
    }
    const float inv = 1.0f / sum;
    for (int t = 0; t < S_; ++t)
        s[(size_t)t*32768 + tid] *= inv;
}

// ---------------------------------------------------------------------------
// Kernel 1c (WMMA): partial wq via LDS-transposed K-chunks.
//   part[wg][m][d] = sum_{r in 1024-row chunk} v[r][m] * q[r][d]
// grid NPART x 128 (4 waves). Per 32-row K-chunk: stage qT[d][r], vT[m][r]
// in LDS (f16), then 16 WMMAs per wave into 16 persistent f32 tile accums.
// Wave w owns mtile = w>>1, dtiles = (w&1)*16 .. +15.
// ---------------------------------------------------------------------------
__global__ __launch_bounds__(128)
void k_wq_wmma(const float* __restrict__ q, const float* __restrict__ v,
               float* __restrict__ part)
{
    __shared__ unsigned short qT[D_*40];   // qT[d][r], stride 40 -> 40960 B
    __shared__ unsigned short vT[M_*40];   // vT[m][r], stride 40 ->  2560 B

    const int tid   = threadIdx.x;
    const int wid   = tid >> 5, lane = tid & 31;
    const int n16   = lane & 15;
    const int hi8   = (lane & 16) ? 8 : 0;
    const int mtile = wid >> 1;                 // 0 or 1
    const int dt0   = (wid & 1) * 16;           // first dtile of this wave
    const long rbase0 = (long)blockIdx.x * 1024;

    const unsigned int* qT32 = (const unsigned int*)qT;   // stride 20 dwords
    const unsigned int* vT32 = (const unsigned int*)vT;

    v8f acc[16];
#pragma unroll
    for (int t = 0; t < 16; ++t) acc[t] = (v8f){};

    for (int ck = 0; ck < 32; ++ck) {
        const long rbase = rbase0 + ck*32;
        // ---- stage q chunk transposed: 32 rows x 512 cols -> qT[d][r] ----
        for (int idx = tid; idx < 32*128; idx += 128) {
            int rr = idx >> 7, d4 = (idx & 127) * 4;
            float4 p = *(const float4*)(q + (size_t)(rbase + rr)*D_ + d4);
            qT[(d4+0)*40 + rr] = h16(p.x);
            qT[(d4+1)*40 + rr] = h16(p.y);
            qT[(d4+2)*40 + rr] = h16(p.z);
            qT[(d4+3)*40 + rr] = h16(p.w);
        }
        // ---- stage v chunk transposed: 32 rows x 32 cols -> vT[m][r] ----
        for (int idx = tid; idx < 32*32; idx += 128) {
            int rr = idx >> 5, m = idx & 31;
            vT[m*40 + rr] = h16(v[(size_t)(rbase + rr)*M_ + m]);
        }
        __syncthreads();

        // ---- one A-fragment per wave (its mtile), 16 B-frags + WMMAs ----
        HFrag a;
#pragma unroll
        for (int j = 0; j < 8; ++j) {
            int ka = a_k(j, lane);
            a.u[j] = vT32[(mtile*16 + n16)*20 + (ka >> 1)];
        }
#pragma unroll
        for (int tt = 0; tt < 16; ++tt) {
            const int dcol = (dt0 + tt)*16 + n16;
            HFrag b;
#pragma unroll
            for (int j = 0; j < 8; ++j) {
                int kb = b_k(j, lane);
                b.u[j] = qT32[dcol*20 + (kb >> 1)];
            }
            acc[tt] = __builtin_amdgcn_wmma_f32_16x16x32_f16(false, a.v, false, b.v, (short)0, acc[tt], false, false);
        }
        __syncthreads();
    }

    // ---- store partial tiles ----
    float* p = part + (size_t)blockIdx.x * (M_*D_);
#pragma unroll
    for (int tt = 0; tt < 16; ++tt) {
        const int dcol = (dt0 + tt)*16 + n16;
#pragma unroll
        for (int i = 0; i < 8; ++i) {
            int m = mtile*16 + i + hi8;
            p[m*D_ + dcol] = acc[tt][i];
        }
    }
}

// ---------------------------------------------------------------------------
// Kernel 1d: wq = (sum of NPART partials) / 1024.  grid 64 x 256.
// ---------------------------------------------------------------------------
__global__ __launch_bounds__(256)
void k_wq_reduce(const float* __restrict__ part, float* __restrict__ wq)
{
    const int tid = blockIdx.x*256 + threadIdx.x;       // 0..16383
    float s = 0.f;
    for (int p = 0; p < NPART; ++p)
        s += part[(size_t)p*(M_*D_) + tid];
    wq[tid] = s * (1.0f/1024.0f);
}

// ---------------------------------------------------------------------------
// Kernel 2: M_new = (1-g)*mem + g*wq, g = sigmoid(mem@U^T + wq@W^T)
// WMMA with chained accumulator. grid 16 x 128 (64 waves = 2 mtiles x 32 dtiles)
// ---------------------------------------------------------------------------
__global__ __launch_bounds__(128)
void k_update(const float* __restrict__ mem, const float* __restrict__ wq,
              const float* __restrict__ Uw, const float* __restrict__ Ww,
              float* __restrict__ Mn)
{
    const int wid   = threadIdx.x >> 5, lane = threadIdx.x & 31;
    const int tile  = blockIdx.x*4 + wid;               // 0..63
    const int mtile = tile >> 5, dtile = tile & 31;
    const int n16   = lane & 15;
    const int hi8   = (lane & 16) ? 8 : 0;
    const int mrow  = mtile*16 + n16;
    const int dcol  = dtile*16 + n16;

    v8f acc = {};
    for (int kc = 0; kc < 16; ++kc) {
        const int kb = kc*32;
        HFrag am, aw, bu, bw;
#pragma unroll
        for (int j = 0; j < 8; ++j) {
            int ka = kb + a_k(j, lane);
            float2 pm = *(const float2*)(mem + (size_t)mrow*D_ + ka);
            float2 pw = *(const float2*)(wq  + (size_t)mrow*D_ + ka);
            am.u[j] = pkh(pm.x, pm.y);
            aw.u[j] = pkh(pw.x, pw.y);
            int kbb = kb + b_k(j, lane);
            float2 pu = *(const float2*)(Uw + (size_t)dcol*D_ + kbb);
            float2 pv = *(const float2*)(Ww + (size_t)dcol*D_ + kbb);
            bu.u[j] = pkh(pu.x, pu.y);
            bw.u[j] = pkh(pv.x, pv.y);
        }
        acc = __builtin_amdgcn_wmma_f32_16x16x32_f16(false, am.v, false, bu.v, (short)0, acc, false, false);
        acc = __builtin_amdgcn_wmma_f32_16x16x32_f16(false, aw.v, false, bw.v, (short)0, acc, false, false);
    }
#pragma unroll
    for (int i = 0; i < 8; ++i) {
        int m = mtile*16 + i + hi8;
        float mm = mem[(size_t)m*D_ + dcol];
        float wv = wq [(size_t)m*D_ + dcol];
        float g  = 1.0f / (1.0f + __expf(-acc[i]));
        Mn[(size_t)m*D_ + dcol] = (1.0f - g)*mm + g*wv;
    }
}

// ---------------------------------------------------------------------------
// Kernel 3 (fused): passthrough copy + Q@M_new^T + softmax over m + w@M_new
// grid 1600 x 128 (4 waves, 16 output rows per wave)
// ---------------------------------------------------------------------------
__global__ __launch_bounds__(128)
void k_out(const float* __restrict__ q, const float* __restrict__ Mn,
           float* __restrict__ out)
{
    __shared__ unsigned short Mb[M_*520];       // M_new as f16, row-major
    __shared__ float          sc[4][16][33];    // per-wave 16x32 scores
    __shared__ unsigned short wl[4][16][40];    // per-wave softmax weights f16

    for (int idx = threadIdx.x; idx < M_*D_; idx += 128) {
        int m = idx >> 9, d = idx & 511;
        Mb[m*520 + d] = h16(Mn[m*D_ + d]);
    }
    __syncthreads();
    const unsigned int* Mb32 = (const unsigned int*)Mb;

    const int wid   = threadIdx.x >> 5, lane = threadIdx.x & 31;
    const int rbase = (blockIdx.x*4 + wid)*16;
    const int n16   = lane & 15;
    const int hi8   = (lane & 16) ? 8 : 0;

    // ---- passthrough: out[r][0:512] = q[r][:] ----
    for (int i = 0; i < 16; ++i) {
        const float4* qs = (const float4*)(q   + (size_t)(rbase + i)*D_);
        float4*       od = (float4*)      (out + (size_t)(rbase + i)*(2*D_));
#pragma unroll
        for (int c = 0; c < 4; ++c)
            od[c*32 + lane] = qs[c*32 + lane];
    }

    // ---- scores2 = Q tile (16x512) @ M_new^T  -> 16x32 ----
    const int arow = rbase + n16;
    v8f acc0 = {}, acc1 = {};
    for (int kc = 0; kc < 16; ++kc) {
        const int kb = kc*32;
        HFrag a, b0, b1;
#pragma unroll
        for (int j = 0; j < 8; ++j) {
            int ka = kb + a_k(j, lane);
            float2 p = *(const float2*)(q + (size_t)arow*D_ + ka);
            a.u[j] = pkh(p.x, p.y);
            int kbb = kb + b_k(j, lane);
            b0.u[j] = Mb32[n16*260        + (kbb >> 1)];
            b1.u[j] = Mb32[(16 + n16)*260 + (kbb >> 1)];
        }
        acc0 = __builtin_amdgcn_wmma_f32_16x16x32_f16(false, a.v, false, b0.v, (short)0, acc0, false, false);
        acc1 = __builtin_amdgcn_wmma_f32_16x16x32_f16(false, a.v, false, b1.v, (short)0, acc1, false, false);
    }
#pragma unroll
    for (int i = 0; i < 8; ++i) {
        sc[wid][i + hi8][n16]      = acc0[i];
        sc[wid][i + hi8][16 + n16] = acc1[i];
    }
    __syncthreads();

    // ---- softmax over m (32) per row; lanes 0..15 handle one row each ----
    if (lane < 16) {
        float mx = -3.0e38f;
#pragma unroll
        for (int m = 0; m < 32; ++m) mx = fmaxf(mx, sc[wid][lane][m]);
        float e[32]; float sum = 0.f;
#pragma unroll
        for (int m = 0; m < 32; ++m) {
            e[m] = __expf((sc[wid][lane][m] - mx) * INV_TEMP);
            sum += e[m];
        }
        const float inv = 1.0f / sum;
#pragma unroll
        for (int m = 0; m < 32; ++m)
            wl[wid][lane][m] = h16(e[m] * inv);
    }
    __syncthreads();

    // ---- retrieved = w (16x32) @ M_new (32x512): one WMMA per 16-wide d tile ----
    HFrag awf;
#pragma unroll
    for (int j = 0; j < 8; ++j) {
        int ka = a_k(j, lane);
        awf.u[j] = *(const unsigned int*)&wl[wid][n16][ka];
    }
    for (int nt = 0; nt < 32; ++nt) {
        const int dcol = nt*16 + n16;
        HFrag bm;
#pragma unroll
        for (int j = 0; j < 8; ++j) {
            int k = b_k(j, lane);
            unsigned int lo = Mb[(size_t)k*520 + dcol];
            unsigned int hi = Mb[(size_t)(k + 1)*520 + dcol];
            bm.u[j] = lo | (hi << 16);
        }
        v8f z = {};
        v8f r = __builtin_amdgcn_wmma_f32_16x16x32_f16(false, awf.v, false, bm.v, (short)0, z, false, false);
#pragma unroll
        for (int i = 0; i < 8; ++i)
            out[(size_t)(rbase + i + hi8)*(2*D_) + D_ + nt*16 + n16] = r[i];
    }
}

// ---------------------------------------------------------------------------
// Launch
// ---------------------------------------------------------------------------
extern "C" void kernel_launch(void* const* d_in, const int* in_sizes, int n_in,
                              void* d_out, int out_size, void* d_ws, size_t ws_size,
                              hipStream_t stream)
{
    (void)in_sizes; (void)n_in; (void)out_size; (void)ws_size;
    const float* q   = (const float*)d_in[0];   // (100,1024,512)
    const float* mem = (const float*)d_in[1];   // (32,512)
    const float* Uw  = (const float*)d_in[2];   // (512,512)
    const float* Ww  = (const float*)d_in[3];   // (512,512)
    float* out = (float*)d_out;                 // (100,1024,1024)
    float* ws  = (float*)d_ws;

    float* scores = ws;                               // 3,276,800 f32 (v_att in place)
    float* part   = scores + (size_t)ROWS*M_;         // NPART * 32 * 512
    float* wq     = part   + (size_t)NPART*M_*D_;     // 16,384
    float* Mn     = wq     + (size_t)M_*D_;           // 16,384

    k_scores    <<<1600,  128, 0, stream>>>(q, mem, scores);
    k_softmax_t <<<128,   256, 0, stream>>>(scores);
    k_wq_wmma   <<<NPART, 128, 0, stream>>>(q, scores, part);
    k_wq_reduce <<<64,    256, 0, stream>>>(part, wq);
    k_update    <<<16,    128, 0, stream>>>(mem, wq, Uw, Ww, Mn);
    k_out       <<<1600,  128, 0, stream>>>(q, Mn, out);
}
